// EfficientMemoryGEMM_20512763806069
// MI455X (gfx1250) — matrix operand order repaired and checked by reading, not compiled, not run on hardware
//
#include <hip/hip_runtime.h>

// ---------------------------------------------------------------------------
// EfficientMemoryGEMM for MI455X (gfx1250, wave32, WMMA)
//   out[0]        : x1 @ x2        (bf16x3-split GEMM on v_wmma_f32_16x16x32_bf16)
//   out[+N]       : x1_hat         (outlier + int8-grid quantize round trip)
//   out[+2N]      : x2_hat         (same pipeline; transpose is a no-op here)
// ---------------------------------------------------------------------------

typedef __attribute__((ext_vector_type(16))) __bf16 v16bf;
typedef __attribute__((ext_vector_type(8)))  float  v8f;

#define SDIM 2048
#define BINS 4096

union FragBF {
  v16bf v;
  uint4 q[2];
};

__device__ __forceinline__ void splitf(float a, __bf16 &h, __bf16 &l) {
  __bf16 hh = (__bf16)a;          // round-to-nearest-even to bf16 (8 mantissa bits)
  h = hh;
  l = (__bf16)(a - (float)hh);    // next ~8 bits
}

// ------------------------------- GEMM --------------------------------------
__global__ __launch_bounds__(256) void k_gemm(const float* __restrict__ A,
                                              const float* __restrict__ B,
                                              float* __restrict__ C)
{
  __shared__ __bf16 As_hi[128][40];   // [M][K], pad to 40 (80B row = 16B aligned, bank-spread)
  __shared__ __bf16 As_lo[128][40];
  __shared__ __bf16 Bs_hi[128][40];   // [N][K] (transposed during staging)
  __shared__ __bf16 Bs_lo[128][40];

  const int tid  = threadIdx.x;
  const int lane = tid & 31;
  const int l16  = lane & 15;
  const int hl   = lane >> 4;          // lane half (0/1)
  const int wave = tid >> 5;           // 0..7
  const int wm   = (wave >> 1) * 32;   // wave M offset inside 128 tile
  const int wn   = (wave & 1) * 64;    // wave N offset inside 128 tile

  const int nBlock = blockIdx.x * 128;
  const int mBlock = blockIdx.y * 128;
  const int bh     = blockIdx.z;

  const float* Abase = A + (size_t)bh * SDIM * SDIM;
  const float* Bbase = B + (size_t)bh * SDIM * SDIM;
  float*       Cbase = C + (size_t)bh * SDIM * SDIM;

  v8f acc[2][4] = {};

  for (int k0 = 0; k0 < SDIM; k0 += 32) {
    // stage A tile: rows mBlock..+127, cols k0..+31 (row-major, float4 coalesced)
#pragma unroll
    for (int t = 0; t < 4; ++t) {
      int i   = tid + t * 256;             // 0..1023 float4s
      int row = i >> 3;                    // 0..127
      int c4  = (i & 7) * 4;               // 0..28
      float4 v = *(const float4*)(Abase + (size_t)(mBlock + row) * SDIM + k0 + c4);
      splitf(v.x, As_hi[row][c4 + 0], As_lo[row][c4 + 0]);
      splitf(v.y, As_hi[row][c4 + 1], As_lo[row][c4 + 1]);
      splitf(v.z, As_hi[row][c4 + 2], As_lo[row][c4 + 2]);
      splitf(v.w, As_hi[row][c4 + 3], As_lo[row][c4 + 3]);
    }
    // stage B tile: global [K][N] -> LDS [N][K] (coalesced read, transposed write)
#pragma unroll
    for (int t = 0; t < 4; ++t) {
      int i  = tid + t * 256;
      int kr = i >> 5;                     // 0..31
      int c4 = (i & 31) * 4;               // 0..124
      float4 v = *(const float4*)(Bbase + (size_t)(k0 + kr) * SDIM + nBlock + c4);
      splitf(v.x, Bs_hi[c4 + 0][kr], Bs_lo[c4 + 0][kr]);
      splitf(v.y, Bs_hi[c4 + 1][kr], Bs_lo[c4 + 1][kr]);
      splitf(v.z, Bs_hi[c4 + 2][kr], Bs_lo[c4 + 2][kr]);
      splitf(v.w, Bs_hi[c4 + 3][kr], Bs_lo[c4 + 3][kr]);
    }
    __syncthreads();

    // A fragment (16x32 bf16): lane half h holds K = [8h..8h+7] then [16+8h..16+8h+7]
    FragBF ahi[2], alo[2];
#pragma unroll
    for (int mi = 0; mi < 2; ++mi) {
      int r = wm + mi * 16 + l16;
      ahi[mi].q[0] = *(const uint4*)&As_hi[r][8 * hl];
      ahi[mi].q[1] = *(const uint4*)&As_hi[r][16 + 8 * hl];
      alo[mi].q[0] = *(const uint4*)&As_lo[r][8 * hl];
      alo[mi].q[1] = *(const uint4*)&As_lo[r][16 + 8 * hl];
    }
    // B fragment (32x16 bf16): lane half h holds K = [8h..8h+7] then [16+8h..16+8h+7] for column N=l16
#pragma unroll
    for (int ni = 0; ni < 4; ++ni) {
      int c = wn + ni * 16 + l16;
      FragBF bhi, blo;
      bhi.q[0] = *(const uint4*)&Bs_hi[c][8 * hl];
      bhi.q[1] = *(const uint4*)&Bs_hi[c][16 + 8 * hl];
      blo.q[0] = *(const uint4*)&Bs_lo[c][8 * hl];
      blo.q[1] = *(const uint4*)&Bs_lo[c][16 + 8 * hl];
#pragma unroll
      for (int mi = 0; mi < 2; ++mi) {
        acc[mi][ni] = __builtin_amdgcn_wmma_f32_16x16x32_bf16(
            false, ahi[mi].v, false, bhi.v, (short)0, acc[mi][ni], false, false);
        acc[mi][ni] = __builtin_amdgcn_wmma_f32_16x16x32_bf16(
            false, alo[mi].v, false, bhi.v, (short)0, acc[mi][ni], false, false);
        acc[mi][ni] = __builtin_amdgcn_wmma_f32_16x16x32_bf16(
            false, ahi[mi].v, false, blo.v, (short)0, acc[mi][ni], false, false);
      }
    }
    __syncthreads();
  }

  // C/D layout: VGPR j, lanes 0-15 -> M=j,N=lane ; lanes 16-31 -> M=j+8,N=lane-16
#pragma unroll
  for (int mi = 0; mi < 2; ++mi)
#pragma unroll
    for (int ni = 0; ni < 4; ++ni) {
      int m0 = mBlock + wm + mi * 16 + 8 * hl;
      int n  = nBlock + wn + ni * 16 + l16;
#pragma unroll
      for (int j = 0; j < 8; ++j)
        Cbase[(size_t)(m0 + j) * SDIM + n] = acc[mi][ni][j];
    }
}

// --------------------------- compression pipeline --------------------------
__global__ void k_zero_hist(unsigned* hist)
{
  int i = blockIdx.x * blockDim.x + threadIdx.x;
  if (i < BINS) hist[i] = 0u;
}

__global__ __launch_bounds__(256) void k_hist1(const float4* __restrict__ x, unsigned n4,
                                               unsigned* __restrict__ hist)
{
  __shared__ unsigned lh[BINS];
  for (int i = threadIdx.x; i < BINS; i += blockDim.x) lh[i] = 0u;
  __syncthreads();
  unsigned idx = blockIdx.x * blockDim.x + threadIdx.x;
  unsigned stride = gridDim.x * blockDim.x;
  for (unsigned i = idx; i < n4; i += stride) {
    float4 v = x[i];
    float a[4] = {fabsf(v.x), fabsf(v.y), fabsf(v.z), fabsf(v.w)};
#pragma unroll
    for (int j = 0; j < 4; ++j) {
      int b = (int)(a[j] * 256.0f);           // bins over [0,16)
      if (b > BINS - 1) b = BINS - 1;
      atomicAdd(&lh[b], 1u);
    }
  }
  __syncthreads();
  for (int i = threadIdx.x; i < BINS; i += blockDim.x)
    if (lh[i]) atomicAdd(&hist[i], lh[i]);
}

__global__ void k_pick1(const unsigned* __restrict__ hist, unsigned k,
                        float* __restrict__ f, unsigned* __restrict__ u)
{
  if (threadIdx.x == 0 && blockIdx.x == 0) {
    unsigned cum = 0; int b = 0;
    for (; b < BINS; ++b) { unsigned c = hist[b]; if (cum + c > k) break; cum += c; }
    if (b == BINS) { b = BINS - 1; cum -= hist[b]; }
    const float w1 = 1.0f / 256.0f;
    f[0] = (float)b * w1;                 // lo of refined range
    f[1] = (float)BINS / w1;              // inv bin width of level 2
    u[0] = k - cum;                       // residual rank
    u[2] = (unsigned)b;                   // level-1 bin id (exact re-binning filter)
  }
}

__global__ __launch_bounds__(256) void k_hist2(const float4* __restrict__ x, unsigned n4,
                                               const float* __restrict__ f,
                                               const unsigned* __restrict__ u,
                                               unsigned* __restrict__ hist)
{
  __shared__ unsigned lh[BINS];
  for (int i = threadIdx.x; i < BINS; i += blockDim.x) lh[i] = 0u;
  __syncthreads();
  float lo = f[0], invw = f[1];
  unsigned sel = u[2];
  unsigned idx = blockIdx.x * blockDim.x + threadIdx.x;
  unsigned stride = gridDim.x * blockDim.x;
  for (unsigned i = idx; i < n4; i += stride) {
    float4 v = x[i];
    float a[4] = {fabsf(v.x), fabsf(v.y), fabsf(v.z), fabsf(v.w)};
#pragma unroll
    for (int j = 0; j < 4; ++j) {
      int b1 = (int)(a[j] * 256.0f);
      if (b1 > BINS - 1) b1 = BINS - 1;
      if ((unsigned)b1 == sel) {
        int b2 = (int)((a[j] - lo) * invw);
        if (b2 < 0) b2 = 0;
        if (b2 > BINS - 1) b2 = BINS - 1;
        atomicAdd(&lh[b2], 1u);
      }
    }
  }
  __syncthreads();
  for (int i = threadIdx.x; i < BINS; i += blockDim.x)
    if (lh[i]) atomicAdd(&hist[i], lh[i]);
}

__global__ void k_pick2(const unsigned* __restrict__ hist, float rfrac,
                        float* __restrict__ f, unsigned* __restrict__ u)
{
  if (threadIdx.x == 0 && blockIdx.x == 0) {
    unsigned k = u[0];
    unsigned cum = 0; int b = 0;
    for (; b < BINS; ++b) { unsigned c = hist[b]; if (cum + c > k) break; cum += c; }
    if (b == BINS) { b = BINS - 1; cum -= hist[b]; }
    unsigned cnt = hist[b]; if (cnt == 0u) cnt = 1u;
    float w2 = 1.0f / f[1];
    float pos = ((float)(k - cum) + rfrac + 0.5f) / (float)cnt;   // uniform-in-bin interp
    if (pos > 1.0f) pos = 1.0f;
    f[2] = f[0] + ((float)b + pos) * w2;   // tau
    u[1] = 0u;                             // init max-|resid| bits
  }
}

__global__ __launch_bounds__(256) void k_maxresid(const float4* __restrict__ x, unsigned n4,
                                                  const float* __restrict__ f,
                                                  unsigned* __restrict__ u)
{
  float tau = f[2];
  unsigned m = 0u;
  unsigned idx = blockIdx.x * blockDim.x + threadIdx.x;
  unsigned stride = gridDim.x * blockDim.x;
  for (unsigned i = idx; i < n4; i += stride) {
    float4 v = x[i];
    float a[4] = {fabsf(v.x), fabsf(v.y), fabsf(v.z), fabsf(v.w)};
#pragma unroll
    for (int j = 0; j < 4; ++j)
      if (a[j] < tau) { unsigned bj = __float_as_uint(a[j]); if (bj > m) m = bj; }
  }
  __shared__ unsigned red[256];
  red[threadIdx.x] = m;
  __syncthreads();
  for (int s = 128; s > 0; s >>= 1) {
    if (threadIdx.x < s) {
      unsigned o = red[threadIdx.x + s];
      if (o > red[threadIdx.x]) red[threadIdx.x] = o;
    }
    __syncthreads();
  }
  if (threadIdx.x == 0) atomicMax(&u[1], red[0]);
}

__device__ __forceinline__ float rec1(float v, float tau, float scale, float inv)
{
  float a = fabsf(v);
  if (a >= tau) return v;                       // exact outlier kept; resid = 0 -> q = 0
  float q = rintf(v * inv);                     // round-half-even like jnp.round
  q = fminf(fmaxf(q, -128.0f), 127.0f);
  return q * scale;
}

__global__ __launch_bounds__(256) void k_rec(const float4* __restrict__ x, unsigned n4,
                                             const float* __restrict__ f,
                                             const unsigned* __restrict__ u,
                                             float4* __restrict__ out)
{
  float tau   = f[2];
  float scale = __uint_as_float(u[1]) * (1.0f / 127.0f);
  float inv   = (scale > 0.0f) ? (1.0f / scale) : 0.0f;
  unsigned idx = blockIdx.x * blockDim.x + threadIdx.x;
  unsigned stride = gridDim.x * blockDim.x;
  for (unsigned i = idx; i < n4; i += stride) {
    float4 v = x[i], o;
    o.x = rec1(v.x, tau, scale, inv);
    o.y = rec1(v.y, tau, scale, inv);
    o.z = rec1(v.z, tau, scale, inv);
    o.w = rec1(v.w, tau, scale, inv);
    out[i] = o;
  }
}

// ------------------------------- launch ------------------------------------
extern "C" void kernel_launch(void* const* d_in, const int* in_sizes, int n_in,
                              void* d_out, int out_size, void* d_ws, size_t ws_size,
                              hipStream_t stream)
{
  const float* x1 = (const float*)d_in[0];   // [2,8,2048,2048] row-major
  const float* x2 = (const float*)d_in[1];   // [2,8,2048,2048] (D x S per head)
  float* out = (float*)d_out;

  const size_t NTOT = (size_t)2 * 8 * 2048 * 2048;   // 67,108,864
  const size_t N0   = NTOT / 2;                      // first batch element
  const unsigned N0_4 = (unsigned)(N0 / 4);
  const unsigned NT_4 = (unsigned)(NTOT / 4);

  unsigned* hist = (unsigned*)d_ws;                                     // 16 KB
  float*    f    = (float*)((char*)d_ws + BINS * sizeof(unsigned));     // 8 floats
  unsigned* u    = (unsigned*)((char*)d_ws + BINS * sizeof(unsigned) + 8 * sizeof(float));

  // 1) batched GEMM (16 heads x 2048^3), 128x128 tile per 256-thread block
  dim3 grid(SDIM / 128, SDIM / 128, 16);
  k_gemm<<<grid, 256, 0, stream>>>(x1, x2, out);

  // 2) compression round trip (quantile rank computed on host, deterministic)
  double r = 0.99 * (double)(N0 - 1);
  unsigned kf = (unsigned)r;
  float rf = (float)(r - (double)kf);

  const float* xs[2]  = {x1, x2};
  float*       xhs[2] = {out + NTOT, out + 2 * NTOT};
  for (int t = 0; t < 2; ++t) {
    k_zero_hist<<<BINS / 256, 256, 0, stream>>>(hist);
    k_hist1<<<2048, 256, 0, stream>>>((const float4*)xs[t], N0_4, hist);
    k_pick1<<<1, 32, 0, stream>>>(hist, kf, f, u);
    k_zero_hist<<<BINS / 256, 256, 0, stream>>>(hist);
    k_hist2<<<2048, 256, 0, stream>>>((const float4*)xs[t], N0_4, f, u, hist);
    k_pick2<<<1, 32, 0, stream>>>(hist, rf, f, u);
    k_maxresid<<<2048, 256, 0, stream>>>((const float4*)xs[t], NT_4, f, u);
    k_rec<<<4096, 256, 0, stream>>>((const float4*)xs[t], NT_4, f, u, (float4*)xhs[t]);
  }
}
